// GlobalCampaignModule_33663953666475
// MI455X (gfx1250) — matrix operand order repaired and checked by reading, not compiled, not run on hardware
//
#include <hip/hip_runtime.h>
#include <hip/hip_bf16.h>

// ---------------- model constants (from reference) ----------------
#define DIMV     255
#define D_MODEL  256
#define D_STATE  64
#define D_CONV   4
#define N_LAYERS 3
#define D_INNER  512        // EXPAND * D_MODEL
#define DT_RANK  16
#define BATCH    32
#define SEQ      2048
#define BL       (BATCH * SEQ)             // 65536 rows, multiple of 128
#define XDIM     (DT_RANK + 2 * D_STATE)   // 144

typedef __attribute__((ext_vector_type(16))) _Float16 v16h;
typedef __attribute__((ext_vector_type(8)))  float    v8f;
typedef int v4i_b128 __attribute__((vector_size(16)));   // b128 payload type for async builtins

#define AS1 __attribute__((address_space(1)))
#define AS3 __attribute__((address_space(3)))

__device__ __forceinline__ float silu_f(float x)     { return x / (1.f + __expf(-x)); }
__device__ __forceinline__ float softplus_f(float x) { return (x > 20.f) ? x : log1pf(__expf(x)); }

// ---- gfx1250 async global->LDS copy (ASYNCcnt path), with safe fallback ----
__device__ __forceinline__ void async_ld_b128(_Float16* lds, const _Float16* g)
{
#if __has_builtin(__builtin_amdgcn_global_load_async_to_lds_b128)
    __builtin_amdgcn_global_load_async_to_lds_b128(
        (AS1 v4i_b128*)(_Float16*)g, (AS3 v4i_b128*)lds, 0, 0);
#else
    *(float4*)lds = *(const float4*)g;
#endif
}
__device__ __forceinline__ void async_wait0()
{
#if __has_builtin(__builtin_amdgcn_s_wait_asynccnt)
    __builtin_amdgcn_s_wait_asynccnt(0);
#endif
}

// ---------------- input build: ti = diff(timestamps), concat -> f16 ----------------
__global__ void build_h0_kernel(const float* __restrict__ auc,
                                const float* __restrict__ ts,
                                _Float16* __restrict__ h0)
{
    size_t idx = (size_t)blockIdx.x * blockDim.x + threadIdx.x;
    if (idx >= (size_t)BL * D_MODEL) return;
    int    dcol = (int)(idx % D_MODEL);
    size_t bl   = idx / D_MODEL;
    int    l    = (int)(bl % SEQ);
    size_t b    = bl / SEQ;
    float v;
    if (dcol < DIMV) v = auc[bl * DIMV + dcol];
    else             v = (l == 0) ? 0.f : ts[b * SEQ + l] - ts[b * SEQ + l - 1];
    h0[idx] = (_Float16)v;
}

// ---------------- f32 -> f16 weight conversion (once per pass) ----------------
__global__ void cvt_f16_kernel(const float* __restrict__ in, _Float16* __restrict__ out, size_t n)
{
    size_t i = (size_t)blockIdx.x * blockDim.x + threadIdx.x;
    if (i < n) out[i] = (_Float16)in[i];
}

// ---------------- A = -exp(A_log), precomputed once (kept f32) ----------------
__global__ void neg_exp_kernel(const float* __restrict__ in, float* __restrict__ out, int n)
{
    int i = blockIdx.x * blockDim.x + threadIdx.x;
    if (i < n) out[i] = -__expf(in[i]);
}

// ---------------- WMMA GEMM: C[M,N](f16) = A[M,K(lda)](f16) * W[N,K](f16)^T ----------------
// Block tile 128x64, 8 waves, each wave = 32x32 via 4x v_wmma_f32_16x16x32_f16.
// K-tiles double-buffered in LDS, staged with async global->LDS b128 copies.
// FULL=true: K%32==0 && N%64==0 -> branch-free staging and stores (in_proj/out_proj).
// ACT: 0 = plain store, 1 = +bias then softplus (dt_proj).
#define BM 128
#define BN 64
#define BK 32
template<bool FULL, int ACT>
__global__ __launch_bounds__(256) void gemm_wmma_f16(
    const _Float16* __restrict__ A, const _Float16* __restrict__ W,
    _Float16* __restrict__ C, int M, int N, int K, int lda,
    const float* __restrict__ bias)
{
    __shared__ _Float16 As[2][BM][BK];   // 2 * 8 KB
    __shared__ _Float16 Ws[2][BN][BK];   // 2 * 4 KB
    const int tid  = threadIdx.x;
    const int lane = tid & 31;
    const int wave = tid >> 5;       // 0..7
    const int wm   = wave & 3;       // 4 sub-tiles of 32 along M
    const int wn   = wave >> 2;      // 2 sub-tiles of 32 along N
    const int m0   = blockIdx.x * BM;
    const int n0   = blockIdx.y * BN;

    v8f acc[2][2] = {};
    const int nk = (K + BK - 1) / BK;

    // stage K-tile t into LDS buffer buf: pure b128 copies (8 f16 per chunk);
    // tails (dt_proj K=16, x_proj N=144) get disjoint ds_store zero-fill.
    auto stage = [&](int t, int buf) {
        const int k0 = t * BK;
        // A: 128 rows x 4 chunks = 512 chunks, 2 per thread
        #pragma unroll
        for (int i = 0; i < (BM * BK) / (256 * 8); ++i) {
            int chunk = i * 256 + tid;
            int r  = chunk >> 2;            // BK/8 = 4 chunks per row
            int c8 = (chunk & 3) * 8;
            int gk = k0 + c8;
            _Float16* dst = &As[buf][r][c8];
            if (FULL || gk < K) {
                async_ld_b128(dst, &A[(size_t)(m0 + r) * lda + gk]);
            } else {
                #pragma unroll
                for (int q = 0; q < 8; ++q) dst[q] = (_Float16)0.f;
            }
        }
        // W: 64 rows x 4 chunks = 256 chunks, 1 per thread
        #pragma unroll
        for (int i = 0; i < (BN * BK) / (256 * 8); ++i) {
            int chunk = i * 256 + tid;
            int r  = chunk >> 2;
            int c8 = (chunk & 3) * 8;
            int gn = n0 + r, gk = k0 + c8;
            _Float16* dst = &Ws[buf][r][c8];
            if (FULL || (gn < N && gk < K)) {
                async_ld_b128(dst, &W[(size_t)gn * K + gk]);
            } else {
                #pragma unroll
                for (int q = 0; q < 8; ++q) dst[q] = (_Float16)0.f;
            }
        }
    };

    stage(0, 0);
    for (int t = 0; t < nk; ++t) {
        async_wait0();        // s_wait_asynccnt 0
        __syncthreads();
        if (t + 1 < nk) stage(t + 1, (t + 1) & 1);
        const int buf  = t & 1;
        const int half = lane >> 4;
        const int r16  = lane & 15;
        // gather fragments per CDNA5 16-bit A(16x32)/B(32x16) lane layouts
        v16h aF[2], bF[2];
        #pragma unroll
        for (int i = 0; i < 2; ++i) {
            #pragma unroll
            for (int p = 0; p < 8; ++p) {
                int kk = (p < 4) ? (half * 8 + 2 * p) : (16 + half * 8 + 2 * (p - 4));
                aF[i][2 * p]     = As[buf][wm * 32 + i * 16 + r16][kk];
                aF[i][2 * p + 1] = As[buf][wm * 32 + i * 16 + r16][kk + 1];
            }
        }
        #pragma unroll
        for (int j = 0; j < 2; ++j) {
            #pragma unroll
            for (int p = 0; p < 8; ++p) {
                int kb = (p < 4) ? (half * 8 + 2 * p) : (16 + half * 8 + 2 * (p - 4));   // lanes 0-15: K=0..7,16..23, lanes 16-31: K=8..15,24..31
                bF[j][2 * p]     = Ws[buf][wn * 32 + j * 16 + r16][kb];
                bF[j][2 * p + 1] = Ws[buf][wn * 32 + j * 16 + r16][kb + 1];
            }
        }
        #pragma unroll
        for (int i = 0; i < 2; ++i)
            #pragma unroll
            for (int j = 0; j < 2; ++j)
                acc[i][j] = __builtin_amdgcn_wmma_f32_16x16x32_f16(
                                false, aF[i], false, bF[j], (short)0, acc[i][j], false, false);
    }

    // f32 C/D layout: lanes 0-15 -> M = v, lanes 16-31 -> M = 8 + v; N = lane&15
    #pragma unroll
    for (int i = 0; i < 2; ++i) {
        #pragma unroll
        for (int j = 0; j < 2; ++j) {
            int ncol = n0 + wn * 32 + j * 16 + (lane & 15);
            if (FULL || ncol < N) {
                float bv = (ACT == 1) ? bias[ncol] : 0.f;
                #pragma unroll
                for (int v = 0; v < 8; ++v) {
                    int mrow = m0 + wm * 32 + i * 16 + (lane >> 4) * 8 + v;
                    float val = acc[i][j][v] + bv;
                    if (ACT == 1) val = softplus_f(val);
                    C[(size_t)mrow * N + ncol] = (_Float16)val;
                }
            }
        }
    }
}

// ---------------- depthwise causal conv-4 + SiLU (f16 in/out, f32 math) ----------------
__global__ void conv_silu_kernel(const _Float16* __restrict__ xz,  // [BL, 1024], xs = cols [0,512)
                                 const float* __restrict__ wc,     // [D_INNER, D_CONV]
                                 const float* __restrict__ bc,     // [D_INNER]
                                 _Float16* __restrict__ xc)        // [BL, D_INNER]
{
    size_t idx = (size_t)blockIdx.x * blockDim.x + threadIdx.x;
    if (idx >= (size_t)BL * D_INNER) return;
    int    d  = (int)(idx % D_INNER);
    size_t bl = idx / D_INNER;
    int    l  = (int)(bl % SEQ);
    float acc = bc[d];
    #pragma unroll
    for (int k = 0; k < D_CONV; ++k) {
        int lk = l + k - (D_CONV - 1);
        if (lk >= 0)
            acc += (float)xz[(bl + (size_t)(lk - l)) * (2 * D_INNER) + d] * wc[d * D_CONV + k];
    }
    xc[idx] = (_Float16)silu_f(acc);
}

// ---------------- selective scan, fused skip + SiLU(z) gate ----------------
// one block per batch, one lane per inner channel; h[64]/A[64] live in VGPRs.
__global__ __launch_bounds__(512) void scan_kernel(
    const _Float16* __restrict__ dtb,   // [BL, D_INNER]  softplus(dt)
    const _Float16* __restrict__ xc,    // [BL, D_INNER]
    const _Float16* __restrict__ dbc,   // [BL, 144]  (B at +16, C at +80)
    const _Float16* __restrict__ xz,    // [BL, 1024] (z at +512)
    const float* __restrict__ aneg,     // [D_INNER, D_STATE]  (-exp(A_log))
    const float* __restrict__ dskip,    // [D_INNER]
    _Float16* __restrict__ yg)          // [BL, D_INNER]
{
    __shared__ float sB[D_STATE], sC[D_STATE];
    const int b = blockIdx.x;
    const int d = threadIdx.x;
    float a[D_STATE], h[D_STATE];
    #pragma unroll
    for (int s = 0; s < D_STATE; ++s) { a[s] = aneg[d * D_STATE + s]; h[s] = 0.f; }
    const float dp = dskip[d];
    for (int l = 0; l < SEQ; ++l) {
        const size_t row = (size_t)b * SEQ + l;
        __syncthreads();
        if (d < 2 * D_STATE) {
            float v = (float)dbc[row * XDIM + DT_RANK + d];
            if (d < D_STATE) sB[d] = v; else sC[d - D_STATE] = v;
        }
        __syncthreads();
        const float dtv = (float)dtb[row * D_INNER + d];
        const float xv  = (float)xc[row * D_INNER + d];
        const float dtx = dtv * xv;
        float y = 0.f;
        #pragma unroll
        for (int s = 0; s < D_STATE; ++s) {
            h[s] = h[s] * __expf(dtv * a[s]) + dtx * sB[s];
            y += h[s] * sC[s];
        }
        const float zv = (float)xz[row * (2 * D_INNER) + D_INNER + d];
        yg[row * D_INNER + d] = (_Float16)((y + xv * dp) * silu_f(zv));
    }
}

// ---------------- tiny MLP head on last timestep ----------------
__global__ __launch_bounds__(128) void mlp_head_kernel(
    const _Float16* __restrict__ hfin,  // [BL, 256]
    const float* __restrict__ accu,     // [B, 3]
    const float* __restrict__ w1, const float* __restrict__ b1,   // [128,259],[128]
    const float* __restrict__ w2, const float* __restrict__ b2,   // [64,128],[64]
    const float* __restrict__ w3, const float* __restrict__ b3,   // [4,64],[4]
    float* __restrict__ out)            // [B, 4]
{
    __shared__ float feat[260];
    __shared__ float y1[128];
    __shared__ float y2[64];
    const int t = threadIdx.x;
    for (int b = 0; b < BATCH; ++b) {
        const size_t row = ((size_t)b * SEQ + (SEQ - 1)) * D_MODEL;
        for (int i = t; i < D_MODEL; i += 128) feat[i] = (float)hfin[row + i];
        if (t < 3) feat[D_MODEL + t] = accu[b * 3 + t];
        __syncthreads();
        {
            float acc = b1[t];
            for (int i = 0; i < DIMV + 4; ++i) acc += feat[i] * w1[t * (DIMV + 4) + i];
            y1[t] = fmaxf(acc, 0.f);
        }
        __syncthreads();
        if (t < 64) {
            float acc = b2[t];
            for (int i = 0; i < 128; ++i) acc += y1[i] * w2[t * 128 + i];
            y2[t] = fmaxf(acc, 0.f);
        }
        __syncthreads();
        if (t < 4) {
            float acc = b3[t];
            for (int i = 0; i < 64; ++i) acc += y2[i] * w3[t * 64 + i];
            out[b * 4 + t] = fmaxf(acc, 0.f);
        }
        __syncthreads();
    }
}

// ---------------- host orchestration ----------------
extern "C" void kernel_launch(void* const* d_in, const int* in_sizes, int n_in,
                              void* d_out, int out_size, void* d_ws, size_t ws_size,
                              hipStream_t stream)
{
    (void)in_sizes; (void)n_in; (void)out_size; (void)ws_size;
    const float* auc  = (const float*)d_in[0];
    const float* ts   = (const float*)d_in[1];
    const float* accu = (const float*)d_in[2];
    const float* Wi   = (const float*)d_in[3];
    const float* Wc   = (const float*)d_in[4];
    const float* bc   = (const float*)d_in[5];
    const float* Wx   = (const float*)d_in[6];
    const float* Wdt  = (const float*)d_in[7];
    const float* bdt  = (const float*)d_in[8];
    const float* Alog = (const float*)d_in[9];
    const float* Dsk  = (const float*)d_in[10];
    const float* Wo   = (const float*)d_in[11];
    const float* w1   = (const float*)d_in[12];
    const float* b1   = (const float*)d_in[13];
    const float* w2   = (const float*)d_in[14];
    const float* b2   = (const float*)d_in[15];
    const float* w3   = (const float*)d_in[16];
    const float* b3   = (const float*)d_in[17];
    float* out = (float*)d_out;

    char* ws = (char*)d_ws;
    size_t off = 0;
    auto alloc = [&](size_t bytes) -> void* {
        void* p = ws + off;
        off += (bytes + 255) & ~(size_t)255;
        return p;
    };
    const size_t nWi = (size_t)N_LAYERS * 2 * D_INNER * D_MODEL;
    const size_t nWx = (size_t)N_LAYERS * XDIM * D_INNER;
    const size_t nWd = (size_t)N_LAYERS * D_INNER * DT_RANK;
    const size_t nWo = (size_t)N_LAYERS * D_MODEL * D_INNER;

    _Float16* hA   = (_Float16*)alloc((size_t)BL * D_MODEL * 2);
    _Float16* hB   = (_Float16*)alloc((size_t)BL * D_MODEL * 2);
    _Float16* xz   = (_Float16*)alloc((size_t)BL * 2 * D_INNER * 2);
    _Float16* xcb  = (_Float16*)alloc((size_t)BL * D_INNER * 2);
    _Float16* dbc  = (_Float16*)alloc((size_t)BL * XDIM * 2);
    _Float16* dtb  = (_Float16*)alloc((size_t)BL * D_INNER * 2);
    _Float16* ygb  = (_Float16*)alloc((size_t)BL * D_INNER * 2);
    float*    aneg = (float*)alloc((size_t)N_LAYERS * D_INNER * D_STATE * 4);
    _Float16* WiH  = (_Float16*)alloc(nWi * 2);
    _Float16* WxH  = (_Float16*)alloc(nWx * 2);
    _Float16* WdH  = (_Float16*)alloc(nWd * 2);
    _Float16* WoH  = (_Float16*)alloc(nWo * 2);

    // one-time prep: input concat, weight conversion, A = -exp(A_log)
    {
        size_t n = (size_t)BL * D_MODEL;
        build_h0_kernel<<<(unsigned)((n + 255) / 256), 256, 0, stream>>>(auc, ts, hA);
    }
    cvt_f16_kernel<<<(unsigned)((nWi + 255) / 256), 256, 0, stream>>>(Wi,  WiH, nWi);
    cvt_f16_kernel<<<(unsigned)((nWx + 255) / 256), 256, 0, stream>>>(Wx,  WxH, nWx);
    cvt_f16_kernel<<<(unsigned)((nWd + 255) / 256), 256, 0, stream>>>(Wdt, WdH, nWd);
    cvt_f16_kernel<<<(unsigned)((nWo + 255) / 256), 256, 0, stream>>>(Wo,  WoH, nWo);
    {
        int n = N_LAYERS * D_INNER * D_STATE;
        neg_exp_kernel<<<(n + 255) / 256, 256, 0, stream>>>(Alog, aneg, n);
    }

    _Float16* hin  = hA;
    _Float16* hnxt = hB;
    for (int l = 0; l < N_LAYERS; ++l) {
        const _Float16* Wi_l  = WiH + (size_t)l * 2 * D_INNER * D_MODEL;
        const float*    Wc_l  = Wc  + (size_t)l * D_INNER * D_CONV;
        const float*    bc_l  = bc  + (size_t)l * D_INNER;
        const _Float16* Wx_l  = WxH + (size_t)l * XDIM * D_INNER;
        const _Float16* Wdt_l = WdH + (size_t)l * D_INNER * DT_RANK;
        const float*    bdt_l = bdt + (size_t)l * D_INNER;
        const float*    Dsk_l = Dsk + (size_t)l * D_INNER;
        const _Float16* Wo_l  = WoH + (size_t)l * D_MODEL * D_INNER;
        const float*    an_l  = aneg + (size_t)l * D_INNER * D_STATE;

        // in_proj: [BL,256] x [1024,256]^T -> xz   (full tiles, branch-free staging)
        dim3 g1(BL / BM, (2 * D_INNER) / BN);
        gemm_wmma_f16<true, 0><<<g1, 256, 0, stream>>>(hin, Wi_l, xz, BL, 2 * D_INNER,
                                                       D_MODEL, D_MODEL, nullptr);
        // conv + silu
        size_t nc = (size_t)BL * D_INNER;
        conv_silu_kernel<<<(unsigned)((nc + 255) / 256), 256, 0, stream>>>(xz, Wc_l, bc_l, xcb);
        // x_proj: [BL,512] x [144,512]^T -> dbc   (N tail zero-filled/guarded)
        dim3 g2(BL / BM, (XDIM + BN - 1) / BN);
        gemm_wmma_f16<false, 0><<<g2, 256, 0, stream>>>(xcb, Wx_l, dbc, BL, XDIM,
                                                        D_INNER, D_INNER, nullptr);
        // dt_proj: dbc[:, :16] x [512,16]^T, bias + softplus -> dtb  (K tail zero-filled)
        dim3 g3(BL / BM, D_INNER / BN);
        gemm_wmma_f16<false, 1><<<g3, 256, 0, stream>>>(dbc, Wdt_l, dtb, BL, D_INNER,
                                                        DT_RANK, XDIM, bdt_l);
        // selective scan (fused skip + gate)
        scan_kernel<<<BATCH, D_INNER, 0, stream>>>(dtb, xcb, dbc, xz, an_l, Dsk_l, ygb);
        // out_proj: [BL,512] x [256,512]^T -> next hidden (full tiles, branch-free staging)
        dim3 g4(BL / BM, D_MODEL / BN);
        gemm_wmma_f16<true, 0><<<g4, 256, 0, stream>>>(ygb, Wo_l, hnxt, BL, D_MODEL,
                                                       D_INNER, D_INNER, nullptr);
        _Float16* tmp = hin; hin = hnxt; hnxt = tmp;
    }

    mlp_head_kernel<<<1, 128, 0, stream>>>(hin, accu, w1, b1, w2, b2, w3, b3, out);
}
